// MultiHeadAttention_39633958207612
// MI455X (gfx1250) — matrix operand order repaired and checked by reading, not compiled, not run on hardware
//
#include <hip/hip_runtime.h>

// ---------------------------------------------------------------------------
// MultiHeadAttention forward for MI455X (gfx1250, wave32, WMMA f16 16x16x32)
// B=4, S=1024, D=1024, H=16, HD=64
// Outputs (concatenated in d_out): out[B,S,D] f32, attn[B,H,S,S] f32
// HBM-write bound (~256MB attn tensor) => fuse energy/softmax/attn*V,
// do every matmul on v_wmma_f32_16x16x32_f16; stage tiles into LDS via
// GLOBAL_LOAD_ASYNC_TO_LDS_B128 where available (ASYNCcnt + s_wait_asynccnt).
// Workspace use: ~56 MB of d_ws (f16 staging buffers).
// ---------------------------------------------------------------------------

#define SEQ     1024
#define DMODEL  1024
#define NHEADS  16
#define HEADD   64
#define BATCH   4
#define BSROWS  (BATCH * SEQ)       // 4096

typedef __attribute__((ext_vector_type(16))) _Float16 v16h;
typedef __attribute__((ext_vector_type(8)))  _Float16 v8h;
typedef __attribute__((ext_vector_type(8)))  float    v8f;

#if defined(__has_builtin)
#if __has_builtin(__builtin_amdgcn_global_load_async_to_lds_b128)
#define HAVE_ASYNC_LDS 1
#endif
#endif
#ifndef HAVE_ASYNC_LDS
#define HAVE_ASYNC_LDS 0
#endif

#if HAVE_ASYNC_LDS
// Builtin signature (from hipcc diagnostic): params are pointers to a 16-byte
// int vector, global (addrspace 1) source and LDS (addrspace 3) destination.
typedef int async_b128_t __attribute__((vector_size(16)));
typedef __attribute__((address_space(1))) async_b128_t* gas_b128_p;
typedef __attribute__((address_space(3))) async_b128_t* las_b128_p;
#endif

// Stage one 16-byte chunk global -> LDS. Async path uses the gfx1250
// async copy (no VGPR round trip, tracked by ASYNCcnt).
__device__ __forceinline__ void stage_b128(const _Float16* __restrict__ g,
                                           _Float16* l) {
#if HAVE_ASYNC_LDS
  __builtin_amdgcn_global_load_async_to_lds_b128((gas_b128_p)g, (las_b128_p)l,
                                                 0, 0);
#else
  *(v8h*)l = *(const v8h*)g;
#endif
}

__device__ __forceinline__ void async_join() {
#if HAVE_ASYNC_LDS
  asm volatile("s_wait_asynccnt 0" ::: "memory");
#endif
}

__device__ __forceinline__ v16h mk16(v8h lo, v8h hi) {
  v16h r;
#pragma unroll
  for (int i = 0; i < 8; ++i) { r[i] = lo[i]; r[i + 8] = hi[i]; }
  return r;
}

__device__ __forceinline__ v8f wmma_f16(v16h a, v16h b, v8f c) {
  // D = A(16x32 f16) * B(32x16 f16) + C(16x16 f32)   [v_wmma_f32_16x16x32_f16]
  return __builtin_amdgcn_wmma_f32_16x16x32_f16(false, a, false, b,
                                                (short)0, c, false, false);
}

// ---------------------------------------------------------------------------
// f32 -> f16 conversion (vectorized)
// ---------------------------------------------------------------------------
__global__ __launch_bounds__(256) void cvt_f32_f16(const float* __restrict__ in,
                                                   _Float16* __restrict__ out,
                                                   int n) {
  int i = (blockIdx.x * 256 + threadIdx.x) * 4;
  if (i < n) {
    float4 v = *(const float4*)(in + i);
    out[i + 0] = (_Float16)v.x;
    out[i + 1] = (_Float16)v.y;
    out[i + 2] = (_Float16)v.z;
    out[i + 3] = (_Float16)v.w;
  }
}

// ---------------------------------------------------------------------------
// GEMM: C[4096 x 1024] = A[4096 x 1024] * W[1024 x 1024] + bias
// Block tile 128x128, 8 waves (one 16-row strip each), K-step 32.
// storeF32 == 0 -> f16 output (Ch), else f32 output (Cf).
// ---------------------------------------------------------------------------
__global__ __launch_bounds__(256) void gemm_wmma(
    const _Float16* __restrict__ A, const _Float16* __restrict__ W,
    const float* __restrict__ bias, _Float16* __restrict__ Ch,
    float* __restrict__ Cf, int storeF32) {
  const int n0   = blockIdx.x * 128;
  const int m0   = blockIdx.y * 128;
  const int tid  = threadIdx.x;
  const int wave = tid >> 5;
  const int lane = tid & 31;
  const int hi   = lane >> 4;     // lane half (0/1)
  const int l16  = lane & 15;

  __shared__ _Float16 As[128][40];   // 128 x 32 tile, stride 40 (16B-aligned rows)
  __shared__ _Float16 Bt[128][40];   // W tile transposed: Bt[n][k]

  v8f acc[8];
#pragma unroll
  for (int i = 0; i < 8; ++i) acc[i] = {};

  const int arow = tid >> 1;
  const int ach  = (tid & 1) * 16;
  const int bk   = tid >> 3;
  const int bnb  = (tid & 7) * 16;

  for (int kb = 0; kb < DMODEL; kb += 32) {
    // --- stage A tile (128 x 32): contiguous 16B chunks -> async to LDS ---
    {
      const _Float16* g = A + (size_t)(m0 + arow) * DMODEL + kb + ach;
      stage_b128(g,     &As[arow][ach]);
      stage_b128(g + 8, &As[arow][ach + 8]);
    }
    // --- stage W tile (32 x 128), transposed into Bt[n][k] (register path) ---
    {
      const v8h* g = (const v8h*)(W + (size_t)(kb + bk) * DMODEL + n0 + bnb);
      v8h t0 = g[0], t1 = g[1];
#pragma unroll
      for (int i = 0; i < 8; ++i) {
        Bt[bnb + i][bk]     = t0[i];
        Bt[bnb + 8 + i][bk] = t1[i];
      }
    }
    async_join();
    __syncthreads();

    // A fragment: lane l16 = row M, elems 0..7 = K hi*8.., 8..15 = K 16+hi*8..
    const int am = wave * 16 + l16;
    v16h af = mk16(*(const v8h*)&As[am][hi * 8],
                   *(const v8h*)&As[am][16 + hi * 8]);
    // Load all 8 B fragments first, then issue the WMMAs back-to-back so the
    // scheduler can batch ds_load_b128s instead of wait-per-wmma.
    v16h bf[8];
#pragma unroll
    for (int nt = 0; nt < 8; ++nt) {
      const int bn = nt * 16 + l16;   // N column
      bf[nt] = mk16(*(const v8h*)&Bt[bn][hi * 8],
                    *(const v8h*)&Bt[bn][16 + hi * 8]);
    }
#pragma unroll
    for (int nt = 0; nt < 8; ++nt) acc[nt] = wmma_f16(af, bf[nt], acc[nt]);
    __syncthreads();
  }

  // Epilogue: C element (m = r + hi*8, n = l16) per VGPR r.  Flag branch is
  // hoisted outside the loops (one branch per thread, not per element).
  if (storeF32) {
#pragma unroll
    for (int nt = 0; nt < 8; ++nt) {
      const int n = n0 + nt * 16 + l16;
      const float bv = bias[n];
#pragma unroll
      for (int r = 0; r < 8; ++r) {
        const int m = m0 + wave * 16 + r + hi * 8;
        Cf[(size_t)m * DMODEL + n] = acc[nt][r] + bv;
      }
    }
  } else {
#pragma unroll
    for (int nt = 0; nt < 8; ++nt) {
      const int n = n0 + nt * 16 + l16;
      const float bv = bias[n];
#pragma unroll
      for (int r = 0; r < 8; ++r) {
        const int m = m0 + wave * 16 + r + hi * 8;
        Ch[(size_t)m * DMODEL + n] = (_Float16)(acc[nt][r] + bv);
      }
    }
  }
}

// ---------------------------------------------------------------------------
// V transpose: Vh[B,S,D] (per-head columns) -> Vt[B*H][HD][SEQ]
// so attn*V B-fragments become contiguous 32B loads.
// ---------------------------------------------------------------------------
__global__ __launch_bounds__(256) void transpose_v(const _Float16* __restrict__ Vh,
                                                   _Float16* __restrict__ Vt) {
  const int bh = blockIdx.y;            // 0..63
  const int b  = bh >> 4, h = bh & 15;
  const int s0 = blockIdx.x * 64;
  __shared__ _Float16 t[64][72];

  {
    int s  = threadIdx.x >> 2;
    int db = (threadIdx.x & 3) * 16;
    const _Float16* g = Vh + ((size_t)(b * SEQ + s0 + s) * DMODEL) + h * HEADD + db;
    stage_b128(g,     &t[s][db]);
    stage_b128(g + 8, &t[s][db + 8]);
  }
  async_join();
  __syncthreads();
  {
    int d  = threadIdx.x >> 2;
    int sb = (threadIdx.x & 3) * 16;
    _Float16* o = Vt + ((size_t)bh * HEADD + d) * SEQ + s0 + sb;
#pragma unroll
    for (int i = 0; i < 16; ++i) o[i] = t[sb + i][d];
  }
}

// ---------------------------------------------------------------------------
// Fused attention: energy (WMMA) -> mask -> softmax -> attn write -> attn*V
// One block per (b, h, 16-query tile). 256 threads = 8 waves.
// LDS: E[16][1028] f32 (65792B) + Qs[16][72] f16 (2304B) + Pred[8][16][16] (8192B)
// ---------------------------------------------------------------------------
#define E_STRIDE   1028
#define E_BYTES    (16 * E_STRIDE * 4)
#define QS_BYTES   (16 * 72 * 2)
#define PRED_BYTES (8 * 16 * 16 * 4)
#define ATTN_SMEM  (E_BYTES + QS_BYTES + PRED_BYTES)

__global__ __launch_bounds__(256) void attn_fused(
    const _Float16* __restrict__ Qh, const _Float16* __restrict__ Kh,
    const _Float16* __restrict__ Vt, const int* __restrict__ mask,
    float* __restrict__ attn_out, _Float16* __restrict__ Oh) {
  extern __shared__ char smem_raw[];
  float    (*E)[E_STRIDE] = (float(*)[E_STRIDE])smem_raw;
  _Float16 (*Qs)[72]      = (_Float16(*)[72])(smem_raw + E_BYTES);
  float    (*Pred)[16][16] = (float(*)[16][16])(smem_raw + E_BYTES + QS_BYTES);

  const int qt = blockIdx.x;            // query tile 0..63
  const int bh = blockIdx.y;            // 0..63
  const int b  = bh >> 4, h = bh & 15;
  const int q0 = qt * 16;
  const int tid  = threadIdx.x;
  const int wave = tid >> 5;
  const int lane = tid & 31;
  const int hi   = lane >> 4;
  const int l16  = lane & 15;

  // ---- stage Q tile (16 x 64): 128 x 16B chunks, async to LDS ----
  if (tid < 128) {
    int r  = tid >> 3;
    int db = (tid & 7) * 8;
    const _Float16* g = Qh + ((size_t)(b * SEQ + q0 + r) * DMODEL) + h * HEADD + db;
    stage_b128(g, &Qs[r][db]);
  }
  async_join();
  __syncthreads();

  // Q A-fragments (reused across all 64 K tiles)
  v16h aq0 = mk16(*(const v8h*)&Qs[l16][hi * 8],      *(const v8h*)&Qs[l16][16 + hi * 8]);
  v16h aq1 = mk16(*(const v8h*)&Qs[l16][32 + hi * 8], *(const v8h*)&Qs[l16][48 + hi * 8]);

  const float inv_sqrt_d = 0.03125f;    // 1/sqrt(1024)
  const float NEG_INF = -__builtin_inff();

  // ---- Phase 1: energy tiles; wave w handles kt = w + 8*i ----
  for (int i = 0; i < 8; ++i) {
    const int kt = wave + i * 8;
    const int k0 = kt * 16;
    const _Float16* kb0 = Kh + ((size_t)(b * SEQ + k0 + l16) * DMODEL) + h * HEADD;
    if (i < 7) {
      // prefetch next K tile -> global_prefetch_b8
      __builtin_prefetch(kb0 + (size_t)128 * DMODEL, 0, 0);
    }
    v16h bk0 = mk16(*(const v8h*)(kb0 + hi * 8),      *(const v8h*)(kb0 + 16 + hi * 8));
    v16h bk1 = mk16(*(const v8h*)(kb0 + 32 + hi * 8), *(const v8h*)(kb0 + 48 + hi * 8));
    v8f acc = {};
    acc = wmma_f16(aq0, bk0, acc);
    acc = wmma_f16(aq1, bk1, acc);
#pragma unroll
    for (int r = 0; r < 8; ++r) {
      const int m = r + hi * 8;
      const int q = q0 + m;
      const int k = k0 + l16;
      E[m][k] = (mask[(size_t)q * SEQ + k] == 0) ? NEG_INF : acc[r] * inv_sqrt_d;
    }
  }
  __syncthreads();

  // ---- Phase 2: row softmax (16 threads per row, shfl_xor width 16) ----
  {
    const int row = tid >> 4;
    const int seg = tid & 15;
    float mx = NEG_INF;
#pragma unroll 4
    for (int j = 0; j < 64; ++j) mx = fmaxf(mx, E[row][seg + 16 * j]);
#pragma unroll
    for (int o = 8; o >= 1; o >>= 1) mx = fmaxf(mx, __shfl_xor(mx, o, 16));
    float sum = 0.f;
#pragma unroll 4
    for (int j = 0; j < 64; ++j) {
      const int c = seg + 16 * j;
      const float v = __expf(E[row][c] - mx);
      E[row][c] = v;
      sum += v;
    }
#pragma unroll
    for (int o = 8; o >= 1; o >>= 1) sum += __shfl_xor(sum, o, 16);
    const float inv = 1.f / sum;
#pragma unroll 4
    for (int j = 0; j < 64; ++j) E[row][seg + 16 * j] *= inv;
  }
  __syncthreads();

  // ---- write attention scores (coalesced, the dominant HBM traffic) ----
  {
    float* ob = attn_out + ((size_t)bh * SEQ + q0) * SEQ;
    for (int idx = tid; idx < 16 * SEQ; idx += 256) {
      const int r = idx >> 10, c = idx & 1023;
      ob[(size_t)r * SEQ + c] = E[r][c];
    }
  }

  // ---- Phase 3: out_head(16x64) = attn(16x1024) @ V, split-K over 2 waves ----
  {
    const int ntile = wave & 3;       // which 16-col slice of HD
    const int khalf = wave >> 2;      // K range half
    v8f acc = {};
    for (int ks = 0; ks < 16; ++ks) {
      const int kb = khalf * 512 + ks * 32;
      // A fragment from post-softmax E (f32 -> f16)
      const float4* pa0 = (const float4*)&E[l16][kb + hi * 8];
      const float4* pa1 = (const float4*)&E[l16][kb + 16 + hi * 8];
      float4 a00 = pa0[0], a01 = pa0[1], a10 = pa1[0], a11 = pa1[1];
      v16h af;
      af[0]  = (_Float16)a00.x; af[1]  = (_Float16)a00.y;
      af[2]  = (_Float16)a00.z; af[3]  = (_Float16)a00.w;
      af[4]  = (_Float16)a01.x; af[5]  = (_Float16)a01.y;
      af[6]  = (_Float16)a01.z; af[7]  = (_Float16)a01.w;
      af[8]  = (_Float16)a10.x; af[9]  = (_Float16)a10.y;
      af[10] = (_Float16)a10.z; af[11] = (_Float16)a10.w;
      af[12] = (_Float16)a11.x; af[13] = (_Float16)a11.y;
      af[14] = (_Float16)a11.z; af[15] = (_Float16)a11.w;
      // B fragment from Vt (contiguous along k)
      const _Float16* vb = Vt + ((size_t)bh * HEADD + ntile * 16 + l16) * SEQ + kb + hi * 8;
      v16h bfv = mk16(*(const v8h*)vb, *(const v8h*)(vb + 16));
      acc = wmma_f16(af, bfv, acc);
    }
#pragma unroll
    for (int r = 0; r < 8; ++r) Pred[wave][r + hi * 8][l16] = acc[r];
  }
  __syncthreads();

  // ---- reduce split-K pairs and store head output (f16, (B,S,D) layout) ----
  for (int idx = tid; idx < 16 * 64; idx += 256) {
    const int m = idx >> 6;
    const int d = idx & 63;
    const int nt = d >> 4, n = d & 15;
    const float v = Pred[nt][m][n] + Pred[nt + 4][m][n];
    Oh[((size_t)(b * SEQ + q0 + m) * DMODEL) + h * HEADD + d] = (_Float16)v;
  }
}

// ---------------------------------------------------------------------------
// Host-side launch
// ---------------------------------------------------------------------------
extern "C" void kernel_launch(void* const* d_in, const int* in_sizes, int n_in,
                              void* d_out, int out_size, void* d_ws, size_t ws_size,
                              hipStream_t stream) {
  const float* z  = (const float*)d_in[0];
  const float* x  = (const float*)d_in[1];
  const int*   mk = (const int*)  d_in[2];
  const float* Wq = (const float*)d_in[3];
  const float* bq = (const float*)d_in[4];
  const float* Wk = (const float*)d_in[5];
  const float* bk = (const float*)d_in[6];
  const float* Wv = (const float*)d_in[7];
  const float* bv = (const float*)d_in[8];
  const float* Wo = (const float*)d_in[9];
  const float* bo = (const float*)d_in[10];

  float* out_f32  = (float*)d_out;                                // [B,S,D]
  float* attn_f32 = (float*)d_out + (size_t)BATCH * SEQ * DMODEL; // [B,H,S,S]

  // Workspace layout (f16 halves); ~56 MB total.
  _Float16* w = (_Float16*)d_ws;
  const size_t ND = (size_t)BSROWS * DMODEL;   // 4M
  const size_t WD = (size_t)DMODEL * DMODEL;   // 1M
  _Float16* xh  = w;            // 4M
  _Float16* zh  = xh  + ND;     // 4M
  _Float16* Wqh = zh  + ND;     // 1M
  _Float16* Wkh = Wqh + WD;
  _Float16* Wvh = Wkh + WD;
  _Float16* Woh = Wvh + WD;
  _Float16* Qh  = Woh + WD;     // 4M
  _Float16* Kh  = Qh  + ND;     // 4M
  _Float16* Vh  = Kh  + ND;     // 4M (consumed by transpose; reused as Oh)
  _Float16* Vt  = Vh  + ND;     // 4M  [B*H][HD][SEQ]
  _Float16* Oh  = Vh;           // alias: Vh dead after transpose_v

  // 1) f32 -> f16 staging
  cvt_f32_f16<<<dim3((int)(ND / 1024)), 256, 0, stream>>>(x,  xh,  (int)ND);
  cvt_f32_f16<<<dim3((int)(ND / 1024)), 256, 0, stream>>>(z,  zh,  (int)ND);
  cvt_f32_f16<<<dim3((int)(WD / 1024)), 256, 0, stream>>>(Wq, Wqh, (int)WD);
  cvt_f32_f16<<<dim3((int)(WD / 1024)), 256, 0, stream>>>(Wk, Wkh, (int)WD);
  cvt_f32_f16<<<dim3((int)(WD / 1024)), 256, 0, stream>>>(Wv, Wvh, (int)WD);
  cvt_f32_f16<<<dim3((int)(WD / 1024)), 256, 0, stream>>>(Wo, Woh, (int)WD);

  // 2) projections (WMMA GEMMs)
  const dim3 ggrid(DMODEL / 128, BSROWS / 128);
  gemm_wmma<<<ggrid, 256, 0, stream>>>(xh, Wqh, bq, Qh, nullptr, 0);
  gemm_wmma<<<ggrid, 256, 0, stream>>>(zh, Wkh, bk, Kh, nullptr, 0);
  gemm_wmma<<<ggrid, 256, 0, stream>>>(zh, Wvh, bv, Vh, nullptr, 0);

  // 3) V -> per-head transposed layout
  transpose_v<<<dim3(SEQ / 64, BATCH * NHEADS), 256, 0, stream>>>(Vh, Vt);

  // 4) fused attention (energy -> softmax -> attn write -> attn*V)
  attn_fused<<<dim3(SEQ / 16, BATCH * NHEADS), 256, ATTN_SMEM, stream>>>(
      Qh, Kh, Vt, mk, attn_f32, Oh);

  // 5) output projection (f32 -> d_out)
  gemm_wmma<<<ggrid, 256, 0, stream>>>(Oh, Woh, bo, nullptr, out_f32, 1);
}
